// CryptoNet_12807592476940
// MI455X (gfx1250) — compile-verified, matrix operand order checked by reading, not run on hardware
//
#include <hip/hip_runtime.h>
#include <hip/hip_bf16.h>
#include <math.h>

typedef __bf16 bf16t;
typedef __attribute__((ext_vector_type(8)))  __bf16 v8bf;
typedef __attribute__((ext_vector_type(16))) __bf16 v16bf;
typedef __attribute__((ext_vector_type(8)))  float  v8f;

#define NWG   64
#define BLK   256
#define B_TOT 1024
#define T_LEN 256
#define I_DIM 128
#define H_BIG 256
#define H_SM  32

// ---------------- workspace layout (bytes) ----------------
static constexpr size_t OFF_CNT  = 0;                                  // barrier counter (256 B)
static constexpr size_t OFF_WIH1 = 256;                                // (1024,128) bf16
static constexpr size_t OFF_WHH1 = OFF_WIH1 + 1024u*128u*2u;           // (1024,256) bf16
static constexpr size_t OFF_WIH2 = OFF_WHH1 + 1024u*256u*2u;           // (1024,256) bf16
static constexpr size_t OFF_WHH2 = OFF_WIH2 + 1024u*256u*2u;           // (1024,256) bf16
static constexpr size_t OFF_WIH3 = OFF_WHH2 + 1024u*256u*2u;           // (128,256)  bf16
static constexpr size_t OFF_WHH3 = OFF_WIH3 + 128u*256u*2u;            // (128,32)   bf16
static constexpr size_t OFF_B1   = OFF_WHH3 + 128u*32u*2u;             // 1024 f32 (bih+bhh)
static constexpr size_t OFF_B2   = OFF_B1 + 1024u*4u;                  // 1024 f32
static constexpr size_t OFF_B3   = OFF_B2 + 1024u*4u;                  // 128 f32 (padded)
static constexpr size_t OFF_PS1  = OFF_B3 + 256u*4u;                   // [NWG][256] f32
static constexpr size_t OFF_PQ1  = OFF_PS1 + (size_t)NWG*256u*4u;
static constexpr size_t OFF_PS2  = OFF_PQ1 + (size_t)NWG*256u*4u;
static constexpr size_t OFF_PQ2  = OFF_PS2 + (size_t)NWG*256u*4u;
static constexpr size_t OFF_PS3  = OFF_PQ2 + (size_t)NWG*256u*4u;      // [NWG][32]
static constexpr size_t OFF_PQ3  = OFF_PS3 + (size_t)NWG*32u*4u;
static constexpr size_t WS_NEED  = OFF_PQ3 + (size_t)NWG*32u*4u;

// ---------------- helpers ----------------
__device__ __forceinline__ float sigmoidf_(float v) { return 1.0f / (1.0f + expf(-v)); }

// Load a 16x32 bf16 A/B fragment slice for this lane: two contiguous 16B chunks.
// lane<16 holds K {0..7,16..23}; lane>=16 holds K {8..15,24..31} (caller adds lk=(lane/16)*8).
__device__ __forceinline__ v16bf ldfrag(const bf16t* p) {
    v8bf lo = *(const v8bf*)p;
    v8bf hi = *(const v8bf*)(p + 16);
    return __builtin_shufflevector(lo, hi, 0,1,2,3,4,5,6,7,8,9,10,11,12,13,14,15);
}

// grid-wide barrier: monotonically increasing target within one launch.
__device__ __forceinline__ void gsync(unsigned* cnt, unsigned target) {
    __threadfence();
    __syncthreads();
    if (threadIdx.x == 0) {
        atomicAdd(cnt, 1u);
        while (atomicAdd(cnt, 0u) < target) { __builtin_amdgcn_s_sleep(2); }
    }
    __syncthreads();
    __threadfence();
}

// One H=256 LSTM layer for this workgroup's 16 batch rows.
// gates[16,1024] = A[16,aK] @ Wih^T + h[16,256] @ Whh^T + bsum.  Wave w owns
// n-tiles q*128 + w*16 (q=0..7), so tile q = 2*gate + jg pairs i/f/g/o per wave.
__device__ __forceinline__ void lstm_layer256(
    const bf16t* __restrict__ aIn, int aK,
    bf16t* shH,                       // recurrent state, read then updated in place
    const bf16t* __restrict__ Wih, const bf16t* __restrict__ Whh,
    const float* __restrict__ bsum,
    float (&c)[2][8], int w, int lm, int lk, int rbase)
{
    v8f zz = {0.f,0.f,0.f,0.f,0.f,0.f,0.f,0.f};
    v8f acc[8];
    #pragma unroll
    for (int q = 0; q < 8; ++q) acc[q] = zz;

    #pragma unroll 1
    for (int kk = 0; kk < aK; kk += 32) {
        v16bf a = ldfrag(aIn + lm * aK + kk + lk);
        #pragma unroll
        for (int q = 0; q < 8; ++q) {
            v16bf b = ldfrag(Wih + (size_t)(q*128 + w*16 + lm) * aK + kk + lk);
            acc[q] = __builtin_amdgcn_wmma_f32_16x16x32_bf16(false, a, false, b,
                                                             (short)0, acc[q], false, false);
        }
    }
    #pragma unroll 1
    for (int kk = 0; kk < H_BIG; kk += 32) {
        v16bf a = ldfrag(shH + lm * H_BIG + kk + lk);
        #pragma unroll
        for (int q = 0; q < 8; ++q) {
            v16bf b = ldfrag(Whh + (size_t)(q*128 + w*16 + lm) * H_BIG + kk + lk);
            acc[q] = __builtin_amdgcn_wmma_f32_16x16x32_bf16(false, a, false, b,
                                                             (short)0, acc[q], false, false);
        }
    }
    __syncthreads();   // all waves done reading shH before anyone rewrites it

    #pragma unroll
    for (int jg = 0; jg < 2; ++jg) {
        int j  = jg*128 + w*16 + lm;          // hidden unit index (column)
        float bI = bsum[j], bF = bsum[256 + j], bG = bsum[512 + j], bO = bsum[768 + j];
        #pragma unroll
        for (int r = 0; r < 8; ++r) {         // D layout: row m = r + rbase, col = lane%16
            float iv = sigmoidf_(acc[jg    ][r] + bI);
            float fv = sigmoidf_(acc[2 + jg][r] + bF);
            float gv = tanhf    (acc[4 + jg][r] + bG);
            float ov = sigmoidf_(acc[6 + jg][r] + bO);
            float cn = fv * c[jg][r] + iv * gv;
            c[jg][r] = cn;
            shH[(r + rbase) * H_BIG + j] = (bf16t)(ov * tanhf(cn));
        }
    }
}

// batch-wide BN (training mode) over all 1024 rows; bf16 in -> bf16 out
__device__ __forceinline__ void bn_block(
    const bf16t* shH, bf16t* shZ, int H,
    float* psum, float* psq, const float* __restrict__ gamma, const float* __restrict__ beta,
    unsigned* cnt, unsigned target, int wg, int tid)
{
    if (tid < H) {
        float s = 0.f, q = 0.f;
        #pragma unroll
        for (int m = 0; m < 16; ++m) { float v = (float)shH[m*H + tid]; s += v; q += v*v; }
        psum[wg*H + tid] = s;
        psq [wg*H + tid] = q;
    }
    gsync(cnt, target);
    if (tid < H) {
        float s = 0.f, q = 0.f;
        for (int g = 0; g < NWG; ++g) { s += psum[g*H + tid]; q += psq[g*H + tid]; }
        float mu  = s * (1.0f / B_TOT);
        float var = q * (1.0f / B_TOT) - mu * mu;
        float sc  = gamma[tid] * rsqrtf(var + 1e-5f);
        float sb  = beta[tid] - mu * sc;
        #pragma unroll
        for (int m = 0; m < 16; ++m)
            shZ[m*H + tid] = (bf16t)((float)shH[m*H + tid] * sc + sb);
    }
    __syncthreads();
}

// ---------------- prep kernels ----------------
__global__ void k_cvt_bf16(const float* __restrict__ s, bf16t* __restrict__ d, int n) {
    int i = blockIdx.x * blockDim.x + threadIdx.x;
    if (i < n) d[i] = (bf16t)s[i];
}
__global__ void k_bias_sum(const float* __restrict__ a, const float* __restrict__ b,
                           float* __restrict__ d, int n) {
    int i = blockIdx.x * blockDim.x + threadIdx.x;
    if (i < n) d[i] = a[i] + b[i];
}

// ---------------- persistent recurrent kernel ----------------
__global__ void __launch_bounds__(BLK, 1) k_lstm_persist(
    const float* __restrict__ x,
    const float* __restrict__ g1, const float* __restrict__ b1,
    const float* __restrict__ g2, const float* __restrict__ b2,
    const float* __restrict__ g3, const float* __restrict__ b3,
    const float* __restrict__ Wl, const float* __restrict__ bl,
    const float* __restrict__ Wl2, const float* __restrict__ bl2,
    char* __restrict__ ws, float* __restrict__ out)
{
    unsigned*     cnt  = (unsigned*)(ws + OFF_CNT);
    const bf16t*  Wih1 = (const bf16t*)(ws + OFF_WIH1);
    const bf16t*  Whh1 = (const bf16t*)(ws + OFF_WHH1);
    const bf16t*  Wih2 = (const bf16t*)(ws + OFF_WIH2);
    const bf16t*  Whh2 = (const bf16t*)(ws + OFF_WHH2);
    const bf16t*  Wih3 = (const bf16t*)(ws + OFF_WIH3);
    const bf16t*  Whh3 = (const bf16t*)(ws + OFF_WHH3);
    const float*  bs1  = (const float*)(ws + OFF_B1);
    const float*  bs2  = (const float*)(ws + OFF_B2);
    const float*  bs3  = (const float*)(ws + OFF_B3);
    float* ps1 = (float*)(ws + OFF_PS1);  float* pq1 = (float*)(ws + OFF_PQ1);
    float* ps2 = (float*)(ws + OFF_PS2);  float* pq2 = (float*)(ws + OFF_PQ2);
    float* ps3 = (float*)(ws + OFF_PS3);  float* pq3 = (float*)(ws + OFF_PQ3);

    __shared__ alignas(16) bf16t sxt[16 * I_DIM];   //  4 KB  x_t tile (bf16)
    __shared__ alignas(16) bf16t sh1[16 * H_BIG];   //  8 KB  h1
    __shared__ alignas(16) bf16t sz1[16 * H_BIG];   //  8 KB  BN(h1)
    __shared__ alignas(16) bf16t sh2[16 * H_BIG];   //  8 KB  h2
    __shared__ alignas(16) bf16t sz2[16 * H_BIG];   //  8 KB  BN(h2)
    __shared__ alignas(16) bf16t sh3[16 * H_SM];    //  1 KB  h3
    __shared__ alignas(16) float sz3[16 * H_SM];    //  2 KB  BN(h3), fp32 for head

    const int tid   = threadIdx.x;
    const int lane  = tid & 31;
    const int w     = tid >> 5;            // wave id 0..7 (wave32)
    const int lm    = lane & 15;           // M (A) / N (B,D) within tile
    const int lk    = (lane >> 4) << 3;    // K sub-offset: 0 or 8
    const int rbase = (lane >> 4) << 3;    // D row offset: 0 or 8
    const int wg    = blockIdx.x;
    const int m0    = wg * 16;             // this WG's batch rows

    // zero recurrent state
    for (int i = tid; i < 16 * H_BIG; i += BLK) { sh1[i] = (bf16t)0.0f; sh2[i] = (bf16t)0.0f; }
    for (int i = tid; i < 16 * H_SM; i += BLK)  sh3[i] = (bf16t)0.0f;
    float c1[2][8], c2[2][8], c3[8];
    #pragma unroll
    for (int a = 0; a < 2; ++a)
        #pragma unroll
        for (int r = 0; r < 8; ++r) { c1[a][r] = 0.f; c2[a][r] = 0.f; }
    #pragma unroll
    for (int r = 0; r < 8; ++r) c3[r] = 0.f;
    __syncthreads();

    unsigned bar = 0;

    for (int t = 0; t < T_LEN; ++t) {
        // ---- stage x_t (fp32 -> bf16 in LDS), prefetch next step ----
        for (int i = tid; i < 16 * I_DIM; i += BLK) {
            int m = i >> 7, f = i & (I_DIM - 1);
            sxt[i] = (bf16t)x[(((size_t)(m0 + m)) * T_LEN + t) * I_DIM + f];
        }
        if (tid < 16 && t + 1 < T_LEN)
            __builtin_prefetch(&x[(((size_t)(m0 + tid)) * T_LEN + (t + 1)) * I_DIM], 0, 1);
        __syncthreads();

        // ---- layer 1 (in=128) ----
        lstm_layer256(sxt, I_DIM, sh1, Wih1, Whh1, bs1, c1, w, lm, lk, rbase);
        __syncthreads();
        bn_block(sh1, sz1, H_BIG, ps1, pq1, g1, b1, cnt, (++bar) * NWG, wg, tid);

        // ---- layer 2 (in=256) ----
        lstm_layer256(sz1, H_BIG, sh2, Wih2, Whh2, bs2, c2, w, lm, lk, rbase);
        __syncthreads();
        bn_block(sh2, sz2, H_BIG, ps2, pq2, g2, b2, cnt, (++bar) * NWG, wg, tid);

        // ---- layer 3 (in=256, H=32): waves 0..1, gate G tile at n = G*32 + w*16 ----
        v8f acc3[4];
        {
            v8f zz = {0.f,0.f,0.f,0.f,0.f,0.f,0.f,0.f};
            #pragma unroll
            for (int G = 0; G < 4; ++G) acc3[G] = zz;
        }
        if (w < 2) {
            #pragma unroll 1
            for (int kk = 0; kk < H_BIG; kk += 32) {
                v16bf a = ldfrag(&sz2[lm * H_BIG + kk + lk]);
                #pragma unroll
                for (int G = 0; G < 4; ++G) {
                    v16bf b = ldfrag(Wih3 + (size_t)(G*32 + w*16 + lm) * H_BIG + kk + lk);
                    acc3[G] = __builtin_amdgcn_wmma_f32_16x16x32_bf16(false, a, false, b,
                                                                      (short)0, acc3[G], false, false);
                }
            }
            {   // recurrent part: K = 32, single WMMA per tile
                v16bf a = ldfrag(&sh3[lm * H_SM + lk]);
                #pragma unroll
                for (int G = 0; G < 4; ++G) {
                    v16bf b = ldfrag(Whh3 + (size_t)(G*32 + w*16 + lm) * H_SM + lk);
                    acc3[G] = __builtin_amdgcn_wmma_f32_16x16x32_bf16(false, a, false, b,
                                                                      (short)0, acc3[G], false, false);
                }
            }
        }
        __syncthreads();   // reads of sh3 complete before update
        if (w < 2) {
            int j = w*16 + lm;
            float bI = bs3[j], bF = bs3[32 + j], bG = bs3[64 + j], bO = bs3[96 + j];
            #pragma unroll
            for (int r = 0; r < 8; ++r) {
                float iv = sigmoidf_(acc3[0][r] + bI);
                float fv = sigmoidf_(acc3[1][r] + bF);
                float gv = tanhf    (acc3[2][r] + bG);
                float ov = sigmoidf_(acc3[3][r] + bO);
                float cn = fv * c3[r] + iv * gv;
                c3[r] = cn;
                sh3[(r + rbase) * H_SM + j] = (bf16t)(ov * tanhf(cn));
            }
        }
        __syncthreads();

        // ---- BN3 (H=32, fp32 output for head) ----
        if (tid < H_SM) {
            float s = 0.f, q = 0.f;
            #pragma unroll
            for (int m = 0; m < 16; ++m) { float v = (float)sh3[m*H_SM + tid]; s += v; q += v*v; }
            ps3[wg*H_SM + tid] = s;  pq3[wg*H_SM + tid] = q;
        }
        gsync(cnt, (++bar) * NWG);
        if (tid < H_SM) {
            float s = 0.f, q = 0.f;
            for (int g = 0; g < NWG; ++g) { s += ps3[g*H_SM + tid]; q += pq3[g*H_SM + tid]; }
            float mu  = s * (1.0f / B_TOT);
            float var = q * (1.0f / B_TOT) - mu * mu;
            float sc  = g3[tid] * rsqrtf(var + 1e-5f);
            float sb  = b3[tid] - mu * sc;
            #pragma unroll
            for (int m = 0; m < 16; ++m) sz3[m*H_SM + tid] = (float)sh3[m*H_SM + tid] * sc + sb;
        }
        __syncthreads();

        // ---- head: relu(z@Wl^T+bl) -> softmax(@Wl2^T+bl2), one thread per row ----
        if (tid < 16) {
            int m = tid;
            float v0 = bl[0], v1 = bl[1];
            #pragma unroll
            for (int j = 0; j < H_SM; ++j) {
                float z = sz3[m*H_SM + j];
                v0 += Wl[j] * z;            // Wl row 0
                v1 += Wl[H_SM + j] * z;     // Wl row 1
            }
            v0 = fmaxf(v0, 0.0f); v1 = fmaxf(v1, 0.0f);
            float u0 = Wl2[0]*v0 + Wl2[1]*v1 + bl2[0];
            float u1 = Wl2[2]*v0 + Wl2[3]*v1 + bl2[1];
            float mx = fmaxf(u0, u1);
            float e0 = expf(u0 - mx), e1 = expf(u1 - mx);
            float inv = 1.0f / (e0 + e1);
            size_t o = (((size_t)(m0 + m)) * T_LEN + t) * 2;
            out[o]     = e0 * inv;
            out[o + 1] = e1 * inv;
        }
        __syncthreads();
    }
}

extern "C" void kernel_launch(void* const* d_in, const int* in_sizes, int n_in,
                              void* d_out, int out_size, void* d_ws, size_t ws_size,
                              hipStream_t stream) {
    const float* x    = (const float*)d_in[0];
    const float* Wih1 = (const float*)d_in[1];
    const float* Whh1 = (const float*)d_in[2];
    const float* bih1 = (const float*)d_in[3];
    const float* bhh1 = (const float*)d_in[4];
    const float* g1   = (const float*)d_in[5];
    const float* b1   = (const float*)d_in[6];
    const float* Wih2 = (const float*)d_in[7];
    const float* Whh2 = (const float*)d_in[8];
    const float* bih2 = (const float*)d_in[9];
    const float* bhh2 = (const float*)d_in[10];
    const float* g2   = (const float*)d_in[11];
    const float* b2   = (const float*)d_in[12];
    const float* Wih3 = (const float*)d_in[13];
    const float* Whh3 = (const float*)d_in[14];
    const float* bih3 = (const float*)d_in[15];
    const float* bhh3 = (const float*)d_in[16];
    const float* g3   = (const float*)d_in[17];
    const float* b3   = (const float*)d_in[18];
    const float* Wl   = (const float*)d_in[19];
    const float* bl   = (const float*)d_in[20];
    const float* Wl2  = (const float*)d_in[21];
    const float* bl2  = (const float*)d_in[22];
    char* ws = (char*)d_ws;

    // deterministic across graph replays: reset grid-barrier counter each call
    hipMemsetAsync(ws + OFF_CNT, 0, 256, stream);

    auto blocks = [](int n) { return (n + 255) / 256; };
    // fp32 -> bf16 weight conversion (natural (N,K) row-major layout)
    k_cvt_bf16<<<blocks(1024*128), 256, 0, stream>>>(Wih1, (bf16t*)(ws + OFF_WIH1), 1024*128);
    k_cvt_bf16<<<blocks(1024*256), 256, 0, stream>>>(Whh1, (bf16t*)(ws + OFF_WHH1), 1024*256);
    k_cvt_bf16<<<blocks(1024*256), 256, 0, stream>>>(Wih2, (bf16t*)(ws + OFF_WIH2), 1024*256);
    k_cvt_bf16<<<blocks(1024*256), 256, 0, stream>>>(Whh2, (bf16t*)(ws + OFF_WHH2), 1024*256);
    k_cvt_bf16<<<blocks(128*256),  256, 0, stream>>>(Wih3, (bf16t*)(ws + OFF_WIH3), 128*256);
    k_cvt_bf16<<<blocks(128*32),   256, 0, stream>>>(Whh3, (bf16t*)(ws + OFF_WHH3), 128*32);
    // combined biases (bih + bhh)
    k_bias_sum<<<blocks(1024), 256, 0, stream>>>(bih1, bhh1, (float*)(ws + OFF_B1), 1024);
    k_bias_sum<<<blocks(1024), 256, 0, stream>>>(bih2, bhh2, (float*)(ws + OFF_B2), 1024);
    k_bias_sum<<<blocks(128),  256, 0, stream>>>(bih3, bhh3, (float*)(ws + OFF_B3), 128);

    // persistent cooperative recurrent kernel: 64 WGs x 16 batch rows, 8 waves each
    k_lstm_persist<<<NWG, BLK, 0, stream>>>(x, g1, b1, g2, b2, g3, b3,
                                            Wl, bl, Wl2, bl2, ws, (float*)d_out);
}
